// BlockAutoregressiveLinear_26869315404606
// MI455X (gfx1250) — matrix operand order repaired and checked by reading, not compiled, not run on hardware
//
#include <hip/hip_runtime.h>
#include <hip/hip_bf16.h>

typedef __attribute__((ext_vector_type(4)))  __bf16 v4bf;
typedef __attribute__((ext_vector_type(8)))  __bf16 v8bf;
typedef __attribute__((ext_vector_type(16))) __bf16 v16bf;
typedef __attribute__((ext_vector_type(8)))  float  v8f;

#define BATCH   2048
#define FEAT    4096
#define YSIZE   ((size_t)BATCH * FEAT)

// ---------------------------------------------------------------------------
// Kernel 1: split x (f32) into bf16 hi/lo planes:  x = hi + lo + O(2^-18 x)
// ---------------------------------------------------------------------------
__global__ void k_split_x(const float* __restrict__ x,
                          __bf16* __restrict__ xh, __bf16* __restrict__ xl) {
  const size_t i = (size_t)blockIdx.x * blockDim.x + threadIdx.x;  // float4 granules
  const float4 v = reinterpret_cast<const float4*>(x)[i];
  float f[4] = {v.x, v.y, v.z, v.w};
  v4bf h, l;
#pragma unroll
  for (int j = 0; j < 4; ++j) {
    __bf16 hb = (__bf16)f[j];
    h[j] = hb;
    l[j] = (__bf16)(f[j] - (float)hb);
  }
  reinterpret_cast<v4bf*>(xh)[i] = h;
  reinterpret_cast<v4bf*>(xl)[i] = l;
}

// ---------------------------------------------------------------------------
// Kernel 2: per output row o:
//   Wm[o,c] = exp(W) on diag 32-block, W on strict-lower blocks, 0 above
//   scale   = exp(W_log_scale[o]) / ||Wm[o,:]||
//   Wn      = Wm * scale  -> bf16 hi/lo planes, zero-padded to 128-aligned cap
//   jac     = log(diag block of Wn) = W + W_log_scale - 0.5*log(sumsq)
// One 256-thread workgroup per row. Mask arrays are never read (structure is
// analytic in the indices).
// ---------------------------------------------------------------------------
__global__ void k_prep_w(const float* __restrict__ W, const float* __restrict__ wls,
                         __bf16* __restrict__ wh, __bf16* __restrict__ wl,
                         float* __restrict__ jac_out) {
  const int o     = blockIdx.x;           // 0..4095
  const int t     = threadIdx.x;          // 0..255
  const int br    = o >> 5;               // 32-block row index
  const int dlo   = br << 5;              // diag block first column
  const int limit = dlo + 32;             // nonzero columns end
  const int cap   = ((o >> 7) + 1) << 7;  // 128-aligned K cap consumed by GEMM
  const float* row = W + (size_t)o * FEAT;

  __shared__ float red[256];
  float ss = 0.f;
  for (int c = t; c < limit; c += 256) {
    float w  = row[c];
    float wm = (c >= dlo) ? __expf(w) : w;
    ss += wm * wm;
  }
  red[t] = ss;
  __syncthreads();
#pragma unroll
  for (int s = 128; s > 0; s >>= 1) {
    if (t < s) red[t] += red[t + s];
    __syncthreads();
  }
  const float sumsq    = red[0];
  const float ls       = wls[o];
  const float scale    = expf(ls) / sqrtf(sumsq);
  const float logscale = ls - 0.5f * logf(sumsq);

  __bf16* whr = wh + (size_t)o * FEAT;
  __bf16* wlr = wl + (size_t)o * FEAT;
  for (int c = t; c < cap; c += 256) {
    float wn = 0.f;
    if (c < limit) {
      float w = row[c];
      if (c >= dlo) {
        wn = expf(w) * scale;
        jac_out[(size_t)br * 1024 + (size_t)(o & 31) * 32 + (c - dlo)] = w + logscale;
      } else {
        wn = w * scale;
      }
    }
    __bf16 hb = (__bf16)wn;
    whr[c] = hb;
    wlr[c] = (__bf16)(wn - (float)hb);
  }
}

// ---------------------------------------------------------------------------
// Kernel 3: y = x @ Wn^T + bias via v_wmma_f32_16x16x32_bf16, 3-term bf16
// split (hi*hi + hi*lo + lo*hi) for ~fp32 accuracy. 128x128 output tile per
// 256-thread (8-wave) workgroup; wave grid 2x4, each wave owns 4x2 16x16
// tiles. K loop capped at blockN+128 (block-lower-triangular Wn).
// ---------------------------------------------------------------------------
__global__ __launch_bounds__(256) void k_gemm(const __bf16* __restrict__ xh,
                                              const __bf16* __restrict__ xl,
                                              const __bf16* __restrict__ wh,
                                              const __bf16* __restrict__ wl,
                                              const float* __restrict__ bias,
                                              float* __restrict__ y) {
  const int lane = threadIdx.x & 31;
  const int wave = threadIdx.x >> 5;
  const int wm   = wave >> 2;          // 0..1 -> 64 batch rows
  const int wn   = wave & 3;           // 0..3 -> 32 out cols
  const int m16  = lane & 15;
  const int half = lane >> 4;

  const int blockM = blockIdx.x * 128;     // batch tile
  const int blockN = blockIdx.y * 128;     // out-feature tile
  const int klim   = blockN + 128;         // triangular K bound

  const int rowBase = blockM + wm * 64 + m16;  // A fragment: this lane's x row
  const int colBase = blockN + wn * 32 + m16;  // B fragment: this lane's Wn row

  v8f acc[4][2];
#pragma unroll
  for (int mt = 0; mt < 4; ++mt)
#pragma unroll
    for (int nt = 0; nt < 2; ++nt)
#pragma unroll
      for (int r = 0; r < 8; ++r) acc[mt][nt][r] = 0.0f;

  for (int k0 = 0; k0 < klim; k0 += 32) {
    v16bf ah[4], al[4], bh[2], bl[2];
    // A fragment: one contiguous 16-elem run per lane: K=[16h,16h+16)
#pragma unroll
    for (int mt = 0; mt < 4; ++mt) {
      const size_t off = (size_t)(rowBase + mt * 16) * FEAT + k0 + 16 * half;
      v8bf h0 = *reinterpret_cast<const v8bf*>(xh + off);
      v8bf h1 = *reinterpret_cast<const v8bf*>(xh + off + 8);
      v8bf l0 = *reinterpret_cast<const v8bf*>(xl + off);
      v8bf l1 = *reinterpret_cast<const v8bf*>(xl + off + 8);
      ah[mt] = __builtin_shufflevector(h0, h1, 0,1,2,3,4,5,6,7,8,9,10,11,12,13,14,15);
      al[mt] = __builtin_shufflevector(l0, l1, 0,1,2,3,4,5,6,7,8,9,10,11,12,13,14,15);
    }
    // B fragment: one contiguous 16-elem run per lane: K=[16h,16h+16)
#pragma unroll
    for (int nt = 0; nt < 2; ++nt) {
      const size_t off = (size_t)(colBase + nt * 16) * FEAT + k0 + 16 * half;
      bh[nt] = *reinterpret_cast<const v16bf*>(wh + off);
      bl[nt] = *reinterpret_cast<const v16bf*>(wl + off);
    }
#pragma unroll
    for (int mt = 0; mt < 4; ++mt)
#pragma unroll
      for (int nt = 0; nt < 2; ++nt) {
        acc[mt][nt] = __builtin_amdgcn_wmma_f32_16x16x32_bf16(
            false, ah[mt], false, bh[nt], (short)0, acc[mt][nt], false, false);
        acc[mt][nt] = __builtin_amdgcn_wmma_f32_16x16x32_bf16(
            false, ah[mt], false, bl[nt], (short)0, acc[mt][nt], false, false);
        acc[mt][nt] = __builtin_amdgcn_wmma_f32_16x16x32_bf16(
            false, al[mt], false, bh[nt], (short)0, acc[mt][nt], false, false);
      }
  }

  float bv[2];
#pragma unroll
  for (int nt = 0; nt < 2; ++nt) bv[nt] = bias[colBase + nt * 16];

  // C/D layout: VGPR r holds M = r + 8*half, N = lane%16
#pragma unroll
  for (int mt = 0; mt < 4; ++mt)
#pragma unroll
    for (int nt = 0; nt < 2; ++nt) {
      const int col = colBase + nt * 16;
#pragma unroll
      for (int r = 0; r < 8; ++r) {
        const int row = blockM + wm * 64 + mt * 16 + half * 8 + r;
        y[(size_t)row * FEAT + col] = acc[mt][nt][r] + bv[nt];
      }
    }
}

// ---------------------------------------------------------------------------
extern "C" void kernel_launch(void* const* d_in, const int* in_sizes, int n_in,
                              void* d_out, int out_size, void* d_ws, size_t ws_size,
                              hipStream_t stream) {
  const float* x    = (const float*)d_in[0];   // (2048, 4096)
  const float* W    = (const float*)d_in[1];   // (4096, 4096)
  const float* bias = (const float*)d_in[2];   // (4096,)
  const float* wls  = (const float*)d_in[3];   // (4096, 1)
  // d_in[4]/d_in[5] (block masks) intentionally unread: structure is analytic.

  float* y   = (float*)d_out;                  // (2048, 4096)
  float* jac = y + YSIZE;                      // (128, 32, 32)

  // Workspace: bf16 hi/lo planes for x and Wn (~96 MiB total)
  __bf16* xh = (__bf16*)d_ws;
  __bf16* xl = xh + (size_t)BATCH * FEAT;
  __bf16* wh = xl + (size_t)BATCH * FEAT;
  __bf16* wl = wh + (size_t)FEAT * FEAT;

  k_split_x<<<(BATCH * FEAT / 4) / 256, 256, 0, stream>>>(x, xh, xl);
  k_prep_w<<<FEAT, 256, 0, stream>>>(W, wls, wh, wl, jac);
  k_gemm<<<dim3(BATCH / 128, FEAT / 128), 256, 0, stream>>>(xh, xl, wh, wl, bias, y);

  (void)in_sizes; (void)n_in; (void)out_size; (void)ws_size;
}
